// ContrastCELoss_1202590842951
// MI455X (gfx1250) — matrix operand order repaired and checked by reading, not compiled, not run on hardware
//
#include <hip/hip_runtime.h>
#include <hip/hip_bf16.h>

// ---------------------------------------------------------------------------
// ContrastCE loss for MI455X (gfx1250), wave32 + WMMA bf16 split-precision GEMM.
//   Pass 0: one-shot fp32 -> (bf16 hi, bf16 lo) plane split of A and B.
//   Pass 1: logits = (Xa @ Xc^T)/0.07 via v_wmma_f32_16x16x32_bf16 (3-term
//           split emulation), fused online (row-max, neg-exp-sum) stats;
//           positive-class logits spilled to workspace.
//   Pass 2: per-row merge + positive log-prob sum.  Pass 3: fixed-order mean.
// ---------------------------------------------------------------------------

typedef __attribute__((ext_vector_type(16))) __bf16 v16bf;
typedef __attribute__((ext_vector_type(8)))  __bf16 v8bf;
typedef __attribute__((ext_vector_type(8)))  float  v8f;

#define FEAT_D     256
#define A_ROWS     1024
#define NUM_C      19
#define CACHE_SZ   5000
#define NCOLS      (NUM_C * CACHE_SZ)        // 95000 contrast columns
#define NREAL      ((NUM_C - 1) * CACHE_SZ)  // 90000 queue-backed columns
#define NPAD       95008                     // columns padded to tile multiple
#define NTILES     5938                      // ceil(95000 / 16)
#define SLICES     32
#define TPS        186                       // ceil(5938 / 32) tiles per slice
#define INV_TEMP   (1.0f / 0.07f)
#define NEG_INF    (-3.0e38f)

#define WMMA_BF16(AH, BH, CC) \
    __builtin_amdgcn_wmma_f32_16x16x32_bf16(false, (AH), false, (BH), (short)0, (CC), false, false)

// ---------------------------------------------------------------------------
// Pass 0a: contrast matrix -> bf16 hi/lo planes, zero-padded to NPAD rows.
// X_contrast[j] == queue_flat[j + 5000] for j < 90000, else zeros.
// Each thread converts 8 contiguous elements (float4 x2 in, 16B x2 out).
// ---------------------------------------------------------------------------
__global__ __launch_bounds__(256)
void prep_contrast(const float* __restrict__ queue,
                   __bf16* __restrict__ chi, __bf16* __restrict__ clo)
{
    const size_t base = ((size_t)blockIdx.x * blockDim.x + threadIdx.x) * 8;
    if (base >= (size_t)NPAD * FEAT_D) return;
    const int j = (int)(base / FEAT_D);

    float x[8];
    if (j < NREAL) {
        const float4* q = reinterpret_cast<const float4*>(queue + base + (size_t)CACHE_SZ * FEAT_D);
        const float4 a = q[0], b = q[1];
        x[0]=a.x; x[1]=a.y; x[2]=a.z; x[3]=a.w;
        x[4]=b.x; x[5]=b.y; x[6]=b.z; x[7]=b.w;
    } else {
        #pragma unroll
        for (int e = 0; e < 8; ++e) x[e] = 0.0f;
    }
    v8bf h, l;
    #pragma unroll
    for (int e = 0; e < 8; ++e) {
        const __bf16 hv = (__bf16)x[e];
        h[e] = hv;
        l[e] = (__bf16)(x[e] - (float)hv);
    }
    *reinterpret_cast<v8bf*>(chi + base) = h;
    *reinterpret_cast<v8bf*>(clo + base) = l;
}

// ---------------------------------------------------------------------------
// Pass 0b: anchors -> bf16 hi/lo planes (1024 x 256, tiny).
// ---------------------------------------------------------------------------
__global__ __launch_bounds__(256)
void prep_anchor(const float* __restrict__ Xa,
                 __bf16* __restrict__ ahi, __bf16* __restrict__ alo)
{
    const size_t base = ((size_t)blockIdx.x * blockDim.x + threadIdx.x) * 8;
    if (base >= (size_t)A_ROWS * FEAT_D) return;
    const float4* q = reinterpret_cast<const float4*>(Xa + base);
    const float4 a = q[0], b = q[1];
    float x[8] = { a.x, a.y, a.z, a.w, b.x, b.y, b.z, b.w };
    v8bf h, l;
    #pragma unroll
    for (int e = 0; e < 8; ++e) {
        const __bf16 hv = (__bf16)x[e];
        h[e] = hv;
        l[e] = (__bf16)(x[e] - (float)hv);
    }
    *reinterpret_cast<v8bf*>(ahi + base) = h;
    *reinterpret_cast<v8bf*>(alo + base) = l;
}

// ---------------------------------------------------------------------------
// Pass 1: WMMA GEMM + fused online softmax stats.  Grid: 64 row-stripes x
// 32 column slices; 4 wave32 per block, each wave owns interleaved 16-col tiles.
// ---------------------------------------------------------------------------
__global__ __launch_bounds__(128, 1)
void contrast_gemm_stats(const __bf16* __restrict__ ahi, const __bf16* __restrict__ alo,
                         const __bf16* __restrict__ chi, const __bf16* __restrict__ clo,
                         const int*   __restrict__ ya,
                         float* __restrict__ pos_ws,   // [1024][5000]
                         float* __restrict__ part_m,   // [1024][SLICES]
                         float* __restrict__ part_s)   // [1024][SLICES]
{
    const int rbase = blockIdx.x * 16;
    const int slice = blockIdx.y;
    const int tid   = threadIdx.x;
    const int wave  = tid >> 5;
    const int lane  = tid & 31;
    const int half  = lane >> 4;
    const int l16   = lane & 15;

    // --- A fragments, resident in VGPRs for the whole stripe ---
    // A 16x32 bf16 layout: lanes -> M=0..15; per half the packed
    // K values are half*16 + {0..15}  (two adjacent 16B runs).
    v16bf a_hi[8], a_lo[8];
    {
        const size_t arow = (size_t)(rbase + l16) * FEAT_D;
        #pragma unroll
        for (int kc = 0; kc < 8; ++kc) {
            const size_t k0 = arow + kc * 32 + half * 16;
            const v8bf h0 = *reinterpret_cast<const v8bf*>(ahi + k0);
            const v8bf h1 = *reinterpret_cast<const v8bf*>(ahi + k0 + 8);
            const v8bf l0 = *reinterpret_cast<const v8bf*>(alo + k0);
            const v8bf l1 = *reinterpret_cast<const v8bf*>(alo + k0 + 8);
            a_hi[kc] = __builtin_shufflevector(h0, h1, 0,1,2,3,4,5,6,7,8,9,10,11,12,13,14,15);
            a_lo[kc] = __builtin_shufflevector(l0, l1, 0,1,2,3,4,5,6,7,8,9,10,11,12,13,14,15);
        }
    }

    int yrow[8];
    #pragma unroll
    for (int r = 0; r < 8; ++r) yrow[r] = ya[rbase + r + 8 * half];

    float m[8], s[8];
    #pragma unroll
    for (int r = 0; r < 8; ++r) { m[r] = NEG_INF; s[r] = 0.0f; }

    const int tstart = slice * TPS;
    const int tend   = (tstart + TPS < NTILES) ? (tstart + TPS) : NTILES;

    for (int t = tstart + wave; t < tend; t += 4) {   // uniform per wave -> EXEC all 1s at WMMA
        const int    j    = t * 16 + l16;             // this lane's contrast column (< NPAD)
        const size_t brow = (size_t)j * FEAT_D;
        if (t + 4 < tend) {                           // prefetch next tile owned by this wave
            __builtin_prefetch(chi + brow + 64 * FEAT_D, 0, 0);
            __builtin_prefetch(clo + brow + 64 * FEAT_D, 0, 0);
        }

        v8f c = {};
        #pragma unroll
        for (int kc = 0; kc < 8; ++kc) {
            // B 32x16 bf16: lanes -> N=0..15; half selects K = half*16 + {0..15}
            // => one contiguous 32B run per plane, loaded straight into v16bf.
            const size_t k0 = brow + kc * 32 + half * 16;
            const v16bf b_hi = *reinterpret_cast<const v16bf*>(chi + k0);
            const v16bf b_lo = *reinterpret_cast<const v16bf*>(clo + k0);
            c = WMMA_BF16(a_hi[kc], b_hi, c);   // hi*hi
            c = WMMA_BF16(a_hi[kc], b_lo, c);   // hi*lo
            c = WMMA_BF16(a_lo[kc], b_hi, c);   // lo*hi  (lo*lo ~2^-16 rel., dropped)
        }

        if (j < NCOLS) {
            const bool jreal = (j < NREAL);
            const int  lbl   = jreal ? (1 + j / CACHE_SZ) : 0;
            const int  pbase = jreal ? (lbl - 1) * CACHE_SZ : NREAL;
            #pragma unroll
            for (int r = 0; r < 8; ++r) {
                const float lg = c[r] * INV_TEMP;
                if (lg > m[r]) {                       // rare: rescale on new row max
                    s[r] *= __expf(m[r] - lg);
                    m[r]  = lg;
                }
                if (lbl != yrow[r]) {                  // negative: accumulate exp
                    s[r] += __expf(lg - m[r]);
                } else {                               // positive: spill raw logit
                    pos_ws[(size_t)(rbase + r + 8 * half) * CACHE_SZ + (j - pbase)] = lg;
                }
            }
        }
    }

    // --- block-level merge (fixed order => deterministic) ---
    __shared__ float red_m[16][64];
    __shared__ float red_s[16][64];
    const int contrib = wave * 16 + l16;
    #pragma unroll
    for (int r = 0; r < 8; ++r) {
        red_m[r + 8 * half][contrib] = m[r];
        red_s[r + 8 * half][contrib] = s[r];
    }
    __syncthreads();
    if (tid < 16) {
        float M = NEG_INF, S = 0.0f;
        for (int k = 0; k < 64; ++k) {
            const float pmv = red_m[tid][k];
            const float psv = red_s[tid][k];
            const float nm  = fmaxf(M, pmv);
            S = S * __expf(M - nm) + psv * __expf(pmv - nm);
            M = nm;
        }
        part_m[(size_t)(rbase + tid) * SLICES + slice] = M;
        part_s[(size_t)(rbase + tid) * SLICES + slice] = S;
    }
}

// ---------------------------------------------------------------------------
// Pass 2: per-row — merge 32 slice partials, then sum log-prob over the 5000
// positive logits (diagonal exclusion only ever hits class-1 rows, since
// contrast columns 0..1023 are all class 1).
// ---------------------------------------------------------------------------
__global__ __launch_bounds__(256)
void contrast_rowloss(const int*   __restrict__ ya,
                      const float* __restrict__ pos_ws,
                      const float* __restrict__ part_m,
                      const float* __restrict__ part_s,
                      float* __restrict__ row_loss)
{
    const int i = blockIdx.x * blockDim.x + threadIdx.x;
    if (i >= A_ROWS) return;

    float M = NEG_INF, S = 0.0f;
    #pragma unroll 4
    for (int sl = 0; sl < SLICES; ++sl) {
        const float pmv = part_m[(size_t)i * SLICES + sl];
        const float psv = part_s[(size_t)i * SLICES + sl];
        const float nm  = fmaxf(M, pmv);
        S = S * __expf(M - nm) + psv * __expf(pmv - nm);
        M = nm;
    }

    const int skip = (ya[i] == 1) ? i : -1;   // self-contrast exclusion
    const float* prow = pos_ws + (size_t)i * CACHE_SZ;
    float acc = 0.0f;
    for (int p = 0; p < CACHE_SZ; ++p) {
        if (p == skip) continue;
        const float t = prow[p] - M;
        acc += t - __logf(__expf(t) + S);     // max-shift cancels exactly
    }
    const float cnt = (float)(CACHE_SZ - (skip >= 0 ? 1 : 0));
    row_loss[i] = -acc / cnt;                 // TEMP/BASE_TEMP == 1
}

// ---------------------------------------------------------------------------
// Pass 3: fixed-order mean over 1024 rows (bitwise deterministic).
// ---------------------------------------------------------------------------
__global__ void contrast_final(const float* __restrict__ row_loss,
                               float* __restrict__ out)
{
    if (blockIdx.x == 0 && threadIdx.x == 0) {
        float acc = 0.0f;
        for (int i = 0; i < A_ROWS; ++i) acc += row_loss[i];
        out[0] = acc * (1.0f / (float)A_ROWS);
    }
}

// ---------------------------------------------------------------------------
extern "C" void kernel_launch(void* const* d_in, const int* in_sizes, int n_in,
                              void* d_out, int out_size, void* d_ws, size_t ws_size,
                              hipStream_t stream) {
    (void)in_sizes; (void)n_in; (void)out_size; (void)ws_size;

    const float* Xa    = (const float*)d_in[0];   // [1024,1,256] fp32
    const int*   ya    = (const int*)  d_in[1];   // [1024] int32
    const float* queue = (const float*)d_in[2];   // [19,5000,256] fp32
    float*       out   = (float*)d_out;           // scalar fp32

    char*  ws  = (char*)d_ws;
    size_t off = 0;
    __bf16* chi = (__bf16*)(ws + off); off += (size_t)NPAD * FEAT_D * sizeof(__bf16);   // 48.6 MB
    __bf16* clo = (__bf16*)(ws + off); off += (size_t)NPAD * FEAT_D * sizeof(__bf16);   // 48.6 MB
    __bf16* ahi = (__bf16*)(ws + off); off += (size_t)A_ROWS * FEAT_D * sizeof(__bf16); // 0.5 MB
    __bf16* alo = (__bf16*)(ws + off); off += (size_t)A_ROWS * FEAT_D * sizeof(__bf16); // 0.5 MB
    float* pos_ws   = (float*)(ws + off); off += (size_t)A_ROWS * CACHE_SZ * sizeof(float); // 20.5 MB
    float* part_m   = (float*)(ws + off); off += (size_t)A_ROWS * SLICES * sizeof(float);
    float* part_s   = (float*)(ws + off); off += (size_t)A_ROWS * SLICES * sizeof(float);
    float* row_loss = (float*)(ws + off);

    prep_contrast<<<dim3((unsigned)((size_t)NPAD * FEAT_D / 8 / 256)), dim3(256), 0, stream>>>(queue, chi, clo);
    prep_anchor<<<dim3((unsigned)((size_t)A_ROWS * FEAT_D / 8 / 256)), dim3(256), 0, stream>>>(Xa, ahi, alo);

    dim3 g1(A_ROWS / 16, SLICES);   // 64 x 32 blocks, 128 threads (4 wave32)
    contrast_gemm_stats<<<g1, dim3(128), 0, stream>>>(ahi, alo, chi, clo, ya, pos_ws, part_m, part_s);
    contrast_rowloss<<<dim3(A_ROWS / 256), dim3(256), 0, stream>>>(ya, pos_ws, part_m, part_s, row_loss);
    contrast_final<<<dim3(1), dim3(64), 0, stream>>>(row_loss, out);
}
